// MultiHeadAttention_72937134621085
// MI455X (gfx1250) — matrix operand order repaired and checked by reading, not compiled, not run on hardware
//
#include <hip/hip_runtime.h>
#include <hip/hip_bf16.h>
#include <math.h>

// ---------------- CDNA5 WMMA types ----------------
typedef __attribute__((ext_vector_type(16))) __bf16 v16bf;
typedef __attribute__((ext_vector_type(8)))  __bf16 v8bf;
typedef __attribute__((ext_vector_type(4)))  __bf16 v4bf;
typedef __attribute__((ext_vector_type(8)))  float  v8f;
typedef __attribute__((ext_vector_type(4)))  float  v4f;

#define D_MODEL   1024
#define NUM_HEADS 16
#define D_K       64
#define BATCH     4
#define SEQ       2048
#define NEGBIG    (-1e20f)

static __device__ __forceinline__ v8f wmma_bf16(v16bf a, v16bf b, v8f c) {
  // D = A(16x32 bf16) * B(32x16 bf16) + C(16x16 f32)
  return __builtin_amdgcn_wmma_f32_16x16x32_bf16(false, a, false, b, (short)0, c,
                                                 false, false);
}

// ---- A-fragment (16x32, MxK) from bf16 row-major [.., ld] -------------------
// ISA 7.12.2: lanes 0-15 row M=lane, elems 0..7 -> K=k0..k0+7, 8..15 -> K=k0+16..23
//             lanes 16-31 same rows, K bases +8.
static __device__ __forceinline__ v16bf load_a_bf16(const __bf16* A, int ld,
                                                    int m0, int k0, int lane) {
  const __bf16* row = A + (size_t)(m0 + (lane & 15)) * ld + k0 + ((lane >> 4) << 3);
  v8bf c0 = *(const v8bf*)(row);
  v8bf c1 = *(const v8bf*)(row + 16);
  v16bf a;
#pragma unroll
  for (int i = 0; i < 8; ++i) { a[i] = c0[i]; a[i + 8] = c1[i]; }
  return a;
}

static __device__ __forceinline__ v16bf load_a_f32(const float* A, int ld,
                                                   int m0, int k0, int lane) {
  const float* row = A + (size_t)(m0 + (lane & 15)) * ld + k0 + ((lane >> 4) << 3);
  v16bf a;
#pragma unroll
  for (int j = 0; j < 2; ++j) {
    v4f c0 = *(const v4f*)(row + 4 * j);
    v4f c1 = *(const v4f*)(row + 16 + 4 * j);
#pragma unroll
    for (int i = 0; i < 4; ++i) {
      a[4 * j + i]     = (__bf16)c0[i];
      a[8 + 4 * j + i] = (__bf16)c1[i];
    }
  }
  return a;
}

// ---- B-fragment (32x16, KxN) where B[k,n] = R[n,k], R row-major stride ld ---
// ISA 7.12.2: lanes 0-15 col N=lane, elems 0..7 -> K=k0..k0+7, 8..15 -> K=k0+16..23; lanes 16-31 K bases +8.
static __device__ __forceinline__ v16bf load_bt_bf16(const __bf16* R, int ld,
                                                     int n0, int k0, int lane) {
  const __bf16* p = R + (size_t)(n0 + (lane & 15)) * ld + k0 + ((lane >> 4) << 3);
  v8bf c0 = *(const v8bf*)(p);
  v8bf c1 = *(const v8bf*)(p + 16);
  v16bf b;
#pragma unroll
  for (int i = 0; i < 8; ++i) { b[i] = c0[i]; b[i + 8] = c1[i]; }
  return b;
}

// ============================================================================
// K0: one-time weight conversion fp32 -> bf16 (W lives in L2; convert once so
// the GEMM inner loops carry zero cvt VALU and half the B-operand bytes).
// ============================================================================
__global__ void cvt_bf16_kernel(const float* __restrict__ src,
                                __bf16* __restrict__ dst, int n) {
  const int i = (blockIdx.x * blockDim.x + threadIdx.x) * 4;
  if (i < n) {
    v4f c = *(const v4f*)(src + i);
    v4bf o;
#pragma unroll
    for (int j = 0; j < 4; ++j) o[j] = (__bf16)c[j];
    *(v4bf*)(dst + i) = o;
  }
}

// ============================================================================
// K1: Y = X @ W^T + b, head-split, bf16 out.  W pre-converted to bf16.
// MODE 0: Y[b,h,s,dk]   (qh, kh)     MODE 1: Y[b,h,dk,s] (vhT, transposed)
// block = 128 thr (4 waves); wave -> 32x64 tile (8 WMMA / k-step, B reused 2x);
// grid (8192/128, 1024/64)
// ============================================================================
template <int MODE>
__global__ void proj_kernel(const float* __restrict__ X, const __bf16* __restrict__ Wb,
                            const float* __restrict__ bias, __bf16* __restrict__ Y) {
  const int wave = threadIdx.x >> 5, lane = threadIdx.x & 31;
  const int m0 = (blockIdx.x * 4 + wave) * 32;
  const int n0 = blockIdx.y * 64;
  v8f acc[2][4] = {};
  for (int k0 = 0; k0 < D_MODEL; k0 += 32) {
    if (k0 + 32 < D_MODEL)
      __builtin_prefetch(X + (size_t)(m0 + (lane & 15)) * D_MODEL + k0 + 32, 0, 0);
    v16bf a0 = load_a_f32(X, D_MODEL, m0,      k0, lane);
    v16bf a1 = load_a_f32(X, D_MODEL, m0 + 16, k0, lane);
#pragma unroll
    for (int t = 0; t < 4; ++t) {
      v16bf b = load_bt_bf16(Wb, D_MODEL, n0 + t * 16, k0, lane);
      acc[0][t] = wmma_bf16(a0, b, acc[0][t]);
      acc[1][t] = wmma_bf16(a1, b, acc[1][t]);
    }
  }
  const int nl = lane & 15, hb = (lane >> 4) * 8;
#pragma unroll
  for (int u = 0; u < 2; ++u) {
#pragma unroll
    for (int t = 0; t < 4; ++t) {
      const int n = n0 + t * 16 + nl;
      const int h = n >> 6, dk = n & 63;
      const float bv = bias[n];
#pragma unroll
      for (int r = 0; r < 8; ++r) {
        const int m = m0 + u * 16 + hb + r;
        const int bb = m >> 11, s = m & (SEQ - 1);
        const float v = acc[u][t][r] + bv;
        size_t idx;
        if (MODE == 0) idx = ((size_t)(bb * NUM_HEADS + h) * SEQ + s) * D_K + dk;
        else           idx = ((size_t)(bb * NUM_HEADS + h) * D_K + dk) * SEQ + s;
        Y[idx] = (__bf16)v;
      }
    }
  }
}

// ============================================================================
// K2: scores = qh @ kh^T / 8, causal mask, softmax -> attn (fp32, HBM once).
// block = 256 thr (8 waves); one 16-query block x all 2048 keys in 128KB LDS.
// grid (SEQ/16, B*H); dynamic LDS = 16*SEQ*4 = 131072 B (<320KB WGP LDS).
// ============================================================================
__global__ void attn_softmax_kernel(const __bf16* __restrict__ qh,
                                    const __bf16* __restrict__ kh,
                                    float* __restrict__ attn) {
  extern __shared__ float sc[];  // [16][SEQ]
  const int wave = threadIdx.x >> 5, lane = threadIdx.x & 31;
  const int qblk = blockIdx.x, bh = blockIdx.y;
  const __bf16* Q = qh + (size_t)bh * SEQ * D_K;
  const __bf16* K = kh + (size_t)bh * SEQ * D_K;
  const int m0 = qblk * 16;

  const v16bf a0 = load_a_bf16(Q, D_K, m0, 0, lane);
  const v16bf a1 = load_a_bf16(Q, D_K, m0, 32, lane);
  const float scale = 0.125f;  // 1/sqrt(64)
  const int nl = lane & 15, hb = (lane >> 4) * 8;

  for (int t = wave; t < SEQ / 16; t += 8) {
    const int n0 = t * 16;
    v16bf b0 = load_bt_bf16(K, D_K, n0, 0, lane);
    v16bf b1 = load_bt_bf16(K, D_K, n0, 32, lane);
    v8f acc = {};
    acc = wmma_bf16(a0, b0, acc);
    acc = wmma_bf16(a1, b1, acc);
    const int n = n0 + nl;
#pragma unroll
    for (int r = 0; r < 8; ++r) {
      const int mrow = hb + r;
      float v = acc[r] * scale;
      if (n > m0 + mrow) v = NEGBIG;  // causal mask
      sc[mrow * SEQ + n] = v;
    }
  }
  __syncthreads();

  // softmax: wave handles rows {wave, wave+8}
#pragma unroll
  for (int rr = 0; rr < 2; ++rr) {
    const int row = wave + rr * 8;
    float* srow = sc + row * SEQ;
    float mx = -INFINITY;
    for (int c = lane; c < SEQ; c += 32) mx = fmaxf(mx, srow[c]);
#pragma unroll
    for (int off = 16; off > 0; off >>= 1) mx = fmaxf(mx, __shfl_xor(mx, off, 32));
    float sum = 0.f;
    for (int c = lane; c < SEQ; c += 32) {
      const float e = __expf(srow[c] - mx);
      srow[c] = e;
      sum += e;
    }
#pragma unroll
    for (int off = 16; off > 0; off >>= 1) sum += __shfl_xor(sum, off, 32);
    const float inv = 1.0f / sum;
    float* out = attn + ((size_t)bh * SEQ + (m0 + row)) * SEQ;
    for (int c = lane * 4; c < SEQ; c += 128) {
      v4f e = *(const v4f*)(srow + c);
      e *= inv;
      *(v4f*)(out + c) = e;
    }
  }
}

// ============================================================================
// K3: out_h = attn @ vh  (K-loop cut at causal boundary; attn==0 beyond).
// block = 128 thr (4 waves); wave -> 16x64 tile; grid (SEQ/64, B*H).
// tmp = combined-heads bf16 [B, S, D_MODEL].
// ============================================================================
__global__ void av_kernel(const float* __restrict__ attn,
                          const __bf16* __restrict__ vhT,
                          __bf16* __restrict__ tmp) {
  const int wave = threadIdx.x >> 5, lane = threadIdx.x & 31;
  const int mtile = blockIdx.x * 4 + wave, bh = blockIdx.y;
  const int m0 = mtile * 16;
  const float* A = attn + (size_t)bh * SEQ * SEQ;
  const __bf16* Vt = vhT + (size_t)bh * D_K * SEQ;  // [64][SEQ]
  v8f acc[4] = {};
  const int kend = m0 + 16;  // causal: attn[m,k]==0 for k>m
  for (int k0 = 0; k0 < kend; k0 += 32) {
    v16bf a = load_a_f32(A, SEQ, m0, k0, lane);
#pragma unroll
    for (int t = 0; t < 4; ++t) {
      v16bf b = load_bt_bf16(Vt, SEQ, t * 16, k0, lane);
      acc[t] = wmma_bf16(a, b, acc[t]);
    }
  }
  const int b = bh >> 4, h = bh & 15;
  const int nl = lane & 15, hb = (lane >> 4) * 8;
#pragma unroll
  for (int t = 0; t < 4; ++t) {
    const int dk = t * 16 + nl;
#pragma unroll
    for (int r = 0; r < 8; ++r) {
      const int s = m0 + hb + r;
      tmp[((size_t)b * SEQ + s) * D_MODEL + h * D_K + dk] = (__bf16)acc[t][r];
    }
  }
}

// ============================================================================
// K4: out = tmp @ Wo^T + bo (fp32 out). 32x64 wave tile like K1, bf16 A and B.
// ============================================================================
__global__ void outproj_kernel(const __bf16* __restrict__ X, const __bf16* __restrict__ Wb,
                               const float* __restrict__ bias, float* __restrict__ Y) {
  const int wave = threadIdx.x >> 5, lane = threadIdx.x & 31;
  const int m0 = (blockIdx.x * 4 + wave) * 32;
  const int n0 = blockIdx.y * 64;
  v8f acc[2][4] = {};
  for (int k0 = 0; k0 < D_MODEL; k0 += 32) {
    v16bf a0 = load_a_bf16(X, D_MODEL, m0,      k0, lane);
    v16bf a1 = load_a_bf16(X, D_MODEL, m0 + 16, k0, lane);
#pragma unroll
    for (int t = 0; t < 4; ++t) {
      v16bf b = load_bt_bf16(Wb, D_MODEL, n0 + t * 16, k0, lane);
      acc[0][t] = wmma_bf16(a0, b, acc[0][t]);
      acc[1][t] = wmma_bf16(a1, b, acc[1][t]);
    }
  }
  const int nl = lane & 15, hb = (lane >> 4) * 8;
#pragma unroll
  for (int u = 0; u < 2; ++u) {
#pragma unroll
    for (int t = 0; t < 4; ++t) {
      const int n = n0 + t * 16 + nl;
      const float bv = bias[n];
#pragma unroll
      for (int r = 0; r < 8; ++r) {
        const int m = m0 + u * 16 + hb + r;
        Y[(size_t)m * D_MODEL + n] = acc[u][t][r] + bv;
      }
    }
  }
}

// ============================================================================
extern "C" void kernel_launch(void* const* d_in, const int* in_sizes, int n_in,
                              void* d_out, int out_size, void* d_ws, size_t ws_size,
                              hipStream_t stream) {
  (void)in_sizes; (void)n_in; (void)out_size; (void)ws_size;
  const float* q    = (const float*)d_in[0];
  const float* k    = (const float*)d_in[1];
  const float* v    = (const float*)d_in[2];
  /* d_in[3] = mask (int32 tril) — causality computed analytically */
  const float* Wq   = (const float*)d_in[4];
  const float* bq   = (const float*)d_in[5];
  const float* Wk   = (const float*)d_in[6];
  const float* bk   = (const float*)d_in[7];
  const float* Wv   = (const float*)d_in[8];
  const float* bv   = (const float*)d_in[9];
  const float* Wo   = (const float*)d_in[10];
  const float* bo   = (const float*)d_in[11];

  // outputs: out [B,S,D_MODEL] fp32, then attn [B,H,S,S] fp32
  float* out  = (float*)d_out;
  float* attn = out + (size_t)BATCH * SEQ * D_MODEL;

  // workspace (bf16): qh | kh | vhT | tmp (16 MiB each) | Wq|Wk|Wv|Wo (2 MiB each)
  const size_t HS = (size_t)BATCH * NUM_HEADS * SEQ * D_K;  // 8,388,608 elems
  const size_t WN = (size_t)D_MODEL * D_MODEL;              // 1,048,576 elems
  __bf16* qh  = (__bf16*)d_ws;
  __bf16* kh  = qh + HS;
  __bf16* vhT = kh + HS;
  __bf16* tmp = vhT + HS;
  __bf16* Wqb = tmp + HS;
  __bf16* Wkb = Wqb + WN;
  __bf16* Wvb = Wkb + WN;
  __bf16* Wob = Wvb + WN;

  // K0: one-time weight fp32->bf16 conversion (stays hot in the 192MB L2)
  {
    const dim3 cb(256), cg((unsigned)(WN / 4 / 256));
    cvt_bf16_kernel<<<cg, cb, 0, stream>>>(Wq, Wqb, (int)WN);
    cvt_bf16_kernel<<<cg, cb, 0, stream>>>(Wk, Wkb, (int)WN);
    cvt_bf16_kernel<<<cg, cb, 0, stream>>>(Wv, Wvb, (int)WN);
    cvt_bf16_kernel<<<cg, cb, 0, stream>>>(Wo, Wob, (int)WN);
  }

  const dim3 blk128(128);
  const dim3 gproj(BATCH * SEQ / 128, D_MODEL / 64);  // (64, 16)

  proj_kernel<0><<<gproj, blk128, 0, stream>>>(q, Wqb, bq, qh);
  proj_kernel<0><<<gproj, blk128, 0, stream>>>(k, Wkb, bk, kh);
  proj_kernel<1><<<gproj, blk128, 0, stream>>>(v, Wvb, bv, vhT);

  const size_t lds_bytes = (size_t)16 * SEQ * sizeof(float);  // 128 KB
  attn_softmax_kernel<<<dim3(SEQ / 16, BATCH * NUM_HEADS), dim3(256), lds_bytes,
                        stream>>>(qh, kh, attn);

  av_kernel<<<dim3(SEQ / 64, BATCH * NUM_HEADS), blk128, 0, stream>>>(attn, vhT, tmp);

  outproj_kernel<<<gproj, blk128, 0, stream>>>(tmp, Wob, bo, out);
}
